// ct_pool_61778809586013
// MI455X (gfx1250) — matrix operand order repaired and checked by reading, not compiled, not run on hardware
//
#include <hip/hip_runtime.h>

#define EPS 1e-3f
#define USE_ASYNC 1

#if defined(__has_builtin)
#if __has_builtin(__builtin_amdgcn_sched_group_barrier)
#define SGB(mask, n) __builtin_amdgcn_sched_group_barrier(mask, n, 0)
#endif
#endif
#ifndef SGB
#define SGB(mask, n)
#endif

typedef __attribute__((ext_vector_type(16))) __bf16 v16bf;
typedef __attribute__((ext_vector_type(8)))  float  v8f;
typedef __attribute__((ext_vector_type(4)))  unsigned int u32x4;
typedef __attribute__((ext_vector_type(2)))  unsigned int u32x2;
typedef __attribute__((ext_vector_type(4)))  float  f32x4;

union Frag16 { u32x4 u[2]; v16bf v; };

typedef __attribute__((address_space(3))) const void as3_cvoid;

// 32-bit LDS byte offset of a generic pointer into a __shared__ array
__device__ __forceinline__ unsigned lds_addr_of(const void* p) {
  return (unsigned)(uintptr_t)(as3_cvoid*)p;
}

__device__ __forceinline__ unsigned short f2bf(float f) {
  unsigned u = __float_as_uint(f);
  u += 0x7FFFu + ((u >> 16) & 1u);            // round-to-nearest-even
  return (unsigned short)(u >> 16);
}
__device__ __forceinline__ float bf2f(unsigned short h) {
  return __uint_as_float(((unsigned)h) << 16);
}

// ---------------- prep kernels ----------------

__global__ __launch_bounds__(256) void cvt_f32_to_bf16(
    const float* __restrict__ x, unsigned short* __restrict__ y, int n4) {
  int i = blockIdx.x * 256 + threadIdx.x;
  if (i >= n4) return;
  f32x4 v = ((const f32x4*)x)[i];
  u32x2 p;
  p.x = (unsigned)f2bf(v.x) | ((unsigned)f2bf(v.y) << 16);
  p.y = (unsigned)f2bf(v.z) | ((unsigned)f2bf(v.w) << 16);
  ((u32x2*)y)[i] = p;
}

// w[Cout][Cin][KK] (f32) -> wpk[k][Cout][Cin] (bf16), ci contiguous
__global__ __launch_bounds__(256) void pack_weights(
    const float* __restrict__ w, unsigned short* __restrict__ out,
    int Cout, int Cin, int KK) {
  int idx = blockIdx.x * 256 + threadIdx.x;
  int total = Cout * Cin * KK;
  if (idx >= total) return;
  int ci = idx % Cin;
  int co = (idx / Cin) % Cout;
  int k  = idx / (Cin * Cout);
  out[idx] = f2bf(w[(co * Cin + ci) * KK + k]);
}

// fold BN: y = conv*s + t with s = g/sqrt(v+eps), t = b - m*s
__global__ __launch_bounds__(256) void bn_prep(
    const float* __restrict__ g, const float* __restrict__ b,
    const float* __restrict__ m, const float* __restrict__ v,
    float* __restrict__ s, float* __restrict__ t, int C) {
  int i = blockIdx.x * 256 + threadIdx.x;
  if (i >= C) return;
  float sc = g[i] * rsqrtf(v[i] + EPS);
  s[i] = sc;
  t[i] = b[i] - m[i] * sc;
}

// S[b,c,h,w] = max_h p1[b,c,:,w] + max_w p2[b,c,h,:]   (double-cummax == global max bcast)
__global__ __launch_bounds__(256) void colrow_max_merge(
    const unsigned short* __restrict__ p1,
    const unsigned short* __restrict__ p2,
    unsigned short* __restrict__ S) {
  const int plane = blockIdx.x;                 // b*128 + c
  const unsigned short* a  = p1 + (size_t)plane * 16384;
  const unsigned short* bm = p2 + (size_t)plane * 16384;
  __shared__ float colmax[128];
  __shared__ float rowmax[128];
  const int tid = threadIdx.x;
  if (tid < 128) {
    float mx = -3.4e38f;
    for (int hh = 0; hh < 128; ++hh) mx = fmaxf(mx, bf2f(a[(hh << 7) + tid]));
    colmax[tid] = mx;
  } else {
    const int hh = tid - 128;
    float mx = -3.4e38f;
    for (int ww = 0; ww < 128; ++ww) mx = fmaxf(mx, bf2f(bm[(hh << 7) + ww]));
    rowmax[hh] = mx;
  }
  __syncthreads();
  unsigned short* so = S + (size_t)plane * 16384;
  for (int i = tid; i < 16384; i += 256)
    so[i] = f2bf(colmax[i & 127] + rowmax[i >> 7]);
}

// ---------------- WMMA implicit-GEMM conv + BN (+skip) (+relu) ----------------
// One workgroup (8 wave32) -> 128 pixels (full W row) x 64 output channels.
// Double-buffered LDS; weights staged via GLOBAL_LOAD_ASYNC_TO_LDS_B128 (ASYNCcnt),
// activations staged as 8-channel gathers + one ds_store_b128 (DS-port friendly);
// one barrier per 32-deep K step; 2-stage register pipeline over taps with
// sched_group_barrier enforcing (10 DS reads | 4 WMMA) interleave.

template<int CIN, int COUT, int KS, bool RELU, bool HAS_SKIP, bool OUT_F32>
__global__ __launch_bounds__(256) void conv_bn_wmma(
    const unsigned short* __restrict__ xbf,     // [B][CIN][128][128] bf16
    const unsigned short* __restrict__ wpk,     // [KK][COUT][CIN] bf16
    const float* __restrict__ scale,
    const float* __restrict__ bias,
    const unsigned short* __restrict__ skip,    // [B][COUT][128][128] bf16 (or unused)
    void* __restrict__ outv) {
  constexpr int KK  = KS * KS;
  constexpr int AW  = (KS == 3) ? 130 : 128;    // haloed pixel count
  constexpr int AST = 40;                       // padded [pixel] row stride (bf16), 80B
  constexpr int BST = 40;
  constexpr int NCO = 64;
  constexpr int Hd = 128, Wd = 128;
  constexpr int HW = Hd * Wd;                   // channel plane stride

  __shared__ __align__(16) unsigned short lds_a[2][KS * AW * AST];
  __shared__ __align__(16) unsigned short lds_b[2][KK * NCO * BST];

  const int co0  = blockIdx.x * NCO;
  const int h    = blockIdx.y;
  const int b    = blockIdx.z;
  const int tid  = threadIdx.x;
  const int lane = tid & 31;
  const int wm   = tid >> 5;
  const int mrow = lane & 15;                   // M row / N col within 16
  const int khalf = lane >> 4;                  // which K half the lane owns

  v8f acc[4] = {};

  // ---- staging helpers ----
  // A: gather 8 consecutive channels (stride H*W) for one haloed pixel and emit
  // a single ds_store_b128 -> [pixel][ci] LDS layout with ~8x fewer DS ops.
  auto stageA = [&](int buf, int ci0) {
    if (KS == 3) {
      for (int i = tid; i < 3 * 4 * 130; i += 256) {
        int w2  = i % 130;                      // haloed pixel, fastest (coalesced)
        int t2  = i / 130;
        int oct = t2 & 3;                       // which 8-channel group
        int kh  = t2 >> 2;
        int hh = h + kh - 1;
        int ww = w2 - 1;
        u32x4 val = {};
        if ((unsigned)hh < (unsigned)Hd && (unsigned)ww < (unsigned)Wd) {
          const unsigned short* g =
              &xbf[((b * CIN + ci0 + oct * 8) * Hd + hh) * Wd + ww];
          unsigned short e[8];
#pragma unroll
          for (int j = 0; j < 8; ++j) e[j] = g[j * HW];
          val.x = (unsigned)e[0] | ((unsigned)e[1] << 16);
          val.y = (unsigned)e[2] | ((unsigned)e[3] << 16);
          val.z = (unsigned)e[4] | ((unsigned)e[5] << 16);
          val.w = (unsigned)e[6] | ((unsigned)e[7] << 16);
        }
        *(u32x4*)&lds_a[buf][(kh * 130 + w2) * AST + oct * 8] = val;
      }
    } else {
      for (int i = tid; i < 4 * 128; i += 256) {
        int w2  = i & 127;
        int oct = i >> 7;
        const unsigned short* g =
            &xbf[((b * CIN + ci0 + oct * 8) * Hd + h) * Wd + w2];
        unsigned short e[8];
#pragma unroll
        for (int j = 0; j < 8; ++j) e[j] = g[j * HW];
        u32x4 val;
        val.x = (unsigned)e[0] | ((unsigned)e[1] << 16);
        val.y = (unsigned)e[2] | ((unsigned)e[3] << 16);
        val.z = (unsigned)e[4] | ((unsigned)e[5] << 16);
        val.w = (unsigned)e[6] | ((unsigned)e[7] << 16);
        *(u32x4*)&lds_a[buf][w2 * AST + oct * 8] = val;
      }
    }
  };

  auto stageB = [&](int buf, int ci0) {
    // KK*NCO*4 is a multiple of 256 -> all lanes always active (async needs EXEC)
    for (int i = tid; i < KK * NCO * 4; i += 256) {
      int ch = i & 3;
      int co = (i >> 2) & (NCO - 1);
      int k  = i >> 8;
      const unsigned short* gsrc =
          &wpk[(k * COUT + co0 + co) * CIN + ci0 + ch * 8];
      unsigned short* ldst = &lds_b[buf][(k * NCO + co) * BST + ch * 8];
#if USE_ASYNC
      unsigned lofs = lds_addr_of(ldst);
      unsigned goff = (unsigned)((const char*)gsrc - (const char*)wpk);
      asm volatile("global_load_async_to_lds_b128 %0, %1, %2"
                   :: "v"(lofs), "v"(goff), "s"(wpk) : "memory");
#else
      *(u32x4*)ldst = *(const u32x4*)gsrc;
#endif
    }
  };

  // fragment fetch (10 ds_load_b128) for one tap into an explicit register set
  auto loadFrags = [&](int cur, int k, Frag16& Af, Frag16 (&Bfr)[4]) {
    const int kh = (KS == 3) ? (k / 3) : 0;
    const int kw = (KS == 3) ? (k % 3) : 0;
    // A frag: lane<16 -> K{0..7,16..23}; lane>=16 -> K{8..15,24..31}
    const int apix = (KS == 3) ? (kh * 130 + wm * 16 + mrow + kw)
                               : (wm * 16 + mrow);
    const unsigned short* ap = &lds_a[cur][apix * AST + khalf * 8];
    Af.u[0] = *(const u32x4*)(ap);
    Af.u[1] = *(const u32x4*)(ap + 16);
#pragma unroll
    for (int nb = 0; nb < 4; ++nb) {
      // B frag: lane<16 -> K{0..7,16..23} of col n; lane>=16 -> K{8..15,24..31}
      const unsigned short* bp =
          &lds_b[cur][(k * NCO + nb * 16 + mrow) * BST + khalf * 8];
      Bfr[nb].u[0] = *(const u32x4*)(bp);
      Bfr[nb].u[1] = *(const u32x4*)(bp + 16);
    }
  };

  // ---- K loop: double-buffered LDS, one barrier per step ----
  constexpr int NSTEP = CIN / 32;
  stageB(0, 0);
  stageA(0, 0);
  for (int s = 0; s < NSTEP; ++s) {
    const int cur = s & 1;
#if USE_ASYNC
    asm volatile("s_wait_asynccnt 0x0" ::: "memory");
#endif
    __syncthreads();
    // get tap-0 fragment loads in flight before staging work is issued
    Frag16 Afr[2];
    Frag16 Bfr[2][4];
    loadFrags(cur, 0, Afr[0], Bfr[0]);
    if (s + 1 < NSTEP) {
      stageB(cur ^ 1, (s + 1) * 32);
      stageA(cur ^ 1, (s + 1) * 32);
    }
    // 2-stage register pipeline over taps: issue tap k+1 loads, then tap k WMMAs
#pragma unroll
    for (int k = 0; k < KK; ++k) {
      const int pb = k & 1;
      if (k + 1 < KK) loadFrags(cur, k + 1, Afr[pb ^ 1], Bfr[pb ^ 1]);
#pragma unroll
      for (int nb = 0; nb < 4; ++nb)
        acc[nb] = __builtin_amdgcn_wmma_f32_16x16x32_bf16(
            false, Afr[pb].v, false, Bfr[pb][nb].v, (short)0, acc[nb],
            false, false);
      // enforce (10 DS reads | 4 WMMA) alternation per tap
      SGB(0x100, 10);   // DS read group
      SGB(0x008, 4);    // WMMA group
    }
  }

  // ---- epilogue: C/D layout -> each lane owns 8 consecutive w for one channel ----
  const int wbase = wm * 16 + khalf * 8;
#pragma unroll
  for (int nb = 0; nb < 4; ++nb) {
    const int co = co0 + nb * 16 + mrow;
    const float s = scale[co];
    const float t = bias[co];
    const int base = (((b * COUT + co) * Hd + h) << 7) + wbase;
    float vals[8];
#pragma unroll
    for (int r = 0; r < 8; ++r) vals[r] = acc[nb][r] * s + t;
    if (HAS_SKIP) {
      u32x4 sk = *(const u32x4*)&skip[base];
      const unsigned short* sp = (const unsigned short*)&sk;
#pragma unroll
      for (int r = 0; r < 8; ++r) vals[r] += bf2f(sp[r]);
    }
    if (RELU) {
#pragma unroll
      for (int r = 0; r < 8; ++r) vals[r] = fmaxf(vals[r], 0.f);
    }
    if (OUT_F32) {
      float* out = (float*)outv;
      f32x4 lo, hi;
      lo.x = vals[0]; lo.y = vals[1]; lo.z = vals[2]; lo.w = vals[3];
      hi.x = vals[4]; hi.y = vals[5]; hi.z = vals[6]; hi.w = vals[7];
      *(f32x4*)&out[base]     = lo;
      *(f32x4*)&out[base + 4] = hi;
    } else {
      unsigned short* out = (unsigned short*)outv;
      u32x4 p;
      p.x = (unsigned)f2bf(vals[0]) | ((unsigned)f2bf(vals[1]) << 16);
      p.y = (unsigned)f2bf(vals[2]) | ((unsigned)f2bf(vals[3]) << 16);
      p.z = (unsigned)f2bf(vals[4]) | ((unsigned)f2bf(vals[5]) << 16);
      p.w = (unsigned)f2bf(vals[6]) | ((unsigned)f2bf(vals[7]) << 16);
      *(u32x4*)&out[base] = p;
    }
  }
}

// ---------------- host orchestration ----------------

extern "C" void kernel_launch(void* const* d_in, const int* in_sizes, int n_in,
                              void* d_out, int out_size, void* d_ws, size_t ws_size,
                              hipStream_t stream) {
  (void)in_sizes; (void)n_in; (void)out_size; (void)ws_size;

  const float* x    = (const float*)d_in[0];
  const float* w_p1 = (const float*)d_in[1];
  const float* g_p1 = (const float*)d_in[2];
  const float* b_p1 = (const float*)d_in[3];
  const float* m_p1 = (const float*)d_in[4];
  const float* v_p1 = (const float*)d_in[5];
  const float* w_p2 = (const float*)d_in[6];
  const float* g_p2 = (const float*)d_in[7];
  const float* b_p2 = (const float*)d_in[8];
  const float* m_p2 = (const float*)d_in[9];
  const float* v_p2 = (const float*)d_in[10];
  const float* w_p  = (const float*)d_in[11];
  const float* g_pb = (const float*)d_in[12];
  const float* b_pb = (const float*)d_in[13];
  const float* m_pb = (const float*)d_in[14];
  const float* v_pb = (const float*)d_in[15];
  const float* w_c1 = (const float*)d_in[16];
  const float* g_b1 = (const float*)d_in[17];
  const float* b_b1 = (const float*)d_in[18];
  const float* m_b1 = (const float*)d_in[19];
  const float* v_b1 = (const float*)d_in[20];
  const float* w_c2 = (const float*)d_in[21];
  const float* g_c2 = (const float*)d_in[22];
  const float* b_c2 = (const float*)d_in[23];
  const float* m_c2 = (const float*)d_in[24];
  const float* v_c2 = (const float*)d_in[25];

  const size_t NX = (size_t)8 * 256 * 128 * 128;   // 33,554,432 elements (C=256)
  const size_t NP = (size_t)8 * 128 * 128 * 128;   // 16,777,216 elements (C=128)

  char* ws = (char*)d_ws;
  size_t off = 0;
  auto take = [&](size_t bytes) -> char* {
    char* p = ws + off;
    off += (bytes + 255) & ~(size_t)255;
    return p;
  };

  unsigned short* xbf   = (unsigned short*)take(NX * 2);
  unsigned short* p1    = (unsigned short*)take(NP * 2);
  unsigned short* p2    = (unsigned short*)take(NP * 2);
  unsigned short* Sbuf  = (unsigned short*)take(NP * 2);
  unsigned short* skipb = (unsigned short*)take(NX * 2);
  unsigned short* relu1 = (unsigned short*)take(NX * 2);
  unsigned short* wk1   = (unsigned short*)take((size_t)9 * 128 * 256 * 2);
  unsigned short* wk2   = (unsigned short*)take((size_t)9 * 128 * 256 * 2);
  unsigned short* wkp   = (unsigned short*)take((size_t)9 * 256 * 128 * 2);
  unsigned short* wkc1  = (unsigned short*)take((size_t)256 * 256 * 2);
  unsigned short* wkc2  = (unsigned short*)take((size_t)9 * 256 * 256 * 2);
  float* sb = (float*)take((size_t)5 * 2 * 256 * 4);
  float* s_p1 = sb + 0 * 256; float* t_p1 = sb + 1 * 256;
  float* s_p2 = sb + 2 * 256; float* t_p2 = sb + 3 * 256;
  float* s_pb = sb + 4 * 256; float* t_pb = sb + 5 * 256;
  float* s_b1 = sb + 6 * 256; float* t_b1 = sb + 7 * 256;
  float* s_c2 = sb + 8 * 256; float* t_c2 = sb + 9 * 256;

  // prep
  cvt_f32_to_bf16<<<dim3((unsigned)((NX / 4 + 255) / 256)), dim3(256), 0, stream>>>(
      x, xbf, (int)(NX / 4));
  pack_weights<<<dim3((9 * 128 * 256 + 255) / 256), dim3(256), 0, stream>>>(w_p1, wk1, 128, 256, 9);
  pack_weights<<<dim3((9 * 128 * 256 + 255) / 256), dim3(256), 0, stream>>>(w_p2, wk2, 128, 256, 9);
  pack_weights<<<dim3((9 * 256 * 128 + 255) / 256), dim3(256), 0, stream>>>(w_p,  wkp, 256, 128, 9);
  pack_weights<<<dim3((256 * 256 + 255) / 256),     dim3(256), 0, stream>>>(w_c1, wkc1, 256, 256, 1);
  pack_weights<<<dim3((9 * 256 * 256 + 255) / 256), dim3(256), 0, stream>>>(w_c2, wkc2, 256, 256, 9);
  bn_prep<<<dim3(1), dim3(256), 0, stream>>>(g_p1, b_p1, m_p1, v_p1, s_p1, t_p1, 128);
  bn_prep<<<dim3(1), dim3(256), 0, stream>>>(g_p2, b_p2, m_p2, v_p2, s_p2, t_p2, 128);
  bn_prep<<<dim3(1), dim3(256), 0, stream>>>(g_pb, b_pb, m_pb, v_pb, s_pb, t_pb, 256);
  bn_prep<<<dim3(1), dim3(256), 0, stream>>>(g_b1, b_b1, m_b1, v_b1, s_b1, t_b1, 256);
  bn_prep<<<dim3(1), dim3(256), 0, stream>>>(g_c2, b_c2, m_c2, v_c2, s_c2, t_c2, 256);

  // branch convs: 3x3 256->128 + BN + relu
  conv_bn_wmma<256, 128, 3, true, false, false>
      <<<dim3(2, 128, 8), dim3(256), 0, stream>>>(xbf, wk1, s_p1, t_p1, nullptr, p1);
  conv_bn_wmma<256, 128, 3, true, false, false>
      <<<dim3(2, 128, 8), dim3(256), 0, stream>>>(xbf, wk2, s_p2, t_p2, nullptr, p2);

  // pool1 + pool2 (global col/row max broadcast) -> S
  colrow_max_merge<<<dim3(8 * 128), dim3(256), 0, stream>>>(p1, p2, Sbuf);

  // skip: 1x1 256->256 + BN (no relu)
  conv_bn_wmma<256, 256, 1, false, false, false>
      <<<dim3(4, 128, 8), dim3(256), 0, stream>>>(xbf, wkc1, s_b1, t_b1, nullptr, skipb);

  // merge: 3x3 128->256 + BN, add skip, relu -> relu1
  conv_bn_wmma<128, 256, 3, true, true, false>
      <<<dim3(4, 128, 8), dim3(256), 0, stream>>>(Sbuf, wkp, s_pb, t_pb, skipb, relu1);

  // output: 3x3 256->256 + BN + relu -> f32 d_out
  conv_bn_wmma<256, 256, 3, true, false, true>
      <<<dim3(4, 128, 8), dim3(256), 0, stream>>>(relu1, wkc2, s_c2, t_c2, nullptr, d_out);
}
